// CrossModalAttentionFusion_3384434229648
// MI455X (gfx1250) — matrix operand order repaired and checked by reading, not compiled, not run on hardware
//
#include <hip/hip_runtime.h>
#include <hip/hip_bf16.h>

// ---------------------------------------------------------------------------
// CrossModalAttentionFusion for MI455X (gfx1250, wave32, WMMA)
// All large GEMMs run on v_wmma_f32_16x16x32_bf16 with LDS double buffering.
// Block tile 128x256x32, 8 waves, wave tile 64x64 (16 WMMAs / K-step).
// ---------------------------------------------------------------------------

typedef __attribute__((ext_vector_type(16))) __bf16 v16bf;
typedef __attribute__((ext_vector_type(8)))  float  v8f;

union FragB16 { v16bf v; uint4 u[2]; };

// ---- bf16 helpers (round-to-nearest-even pack, cheap unpack) --------------
__device__ __forceinline__ unsigned short f2bf(float f) {
    unsigned u = __float_as_uint(f);
    unsigned r = ((u >> 16) & 1u) + 0x7fffu;
    return (unsigned short)((u + r) >> 16);
}
__device__ __forceinline__ float bf2f(unsigned short h) {
    return __uint_as_float((unsigned)h << 16);
}
__device__ __forceinline__ unsigned int packbf2(float a, float b) {
    return (unsigned)f2bf(a) | ((unsigned)f2bf(b) << 16);
}
__device__ __forceinline__ float gelu_exact(float x) {
    return 0.5f * x * (1.0f + erff(x * 0.70710678118654752440f));
}

// ---- global fetch of 16 elements (bf16-packed, two uint4) -----------------
__device__ __forceinline__ void fetch16(const float* __restrict__ s, uint4& d0, uint4& d1) {
    const float4* p = (const float4*)s;
    float4 f0 = p[0], f1 = p[1], f2 = p[2], f3 = p[3];
    d0.x = packbf2(f0.x, f0.y); d0.y = packbf2(f0.z, f0.w);
    d0.z = packbf2(f1.x, f1.y); d0.w = packbf2(f1.z, f1.w);
    d1.x = packbf2(f2.x, f2.y); d1.y = packbf2(f2.z, f2.w);
    d1.z = packbf2(f3.x, f3.y); d1.w = packbf2(f3.z, f3.w);
}
__device__ __forceinline__ void fetch16(const unsigned short* __restrict__ s, uint4& d0, uint4& d1) {
    const uint4* p = (const uint4*)s;
    d0 = p[0]; d1 = p[1];
}
// 32 bf16 elements (one full K-chunk row of the weight tile)
__device__ __forceinline__ void fetchW32(const unsigned short* __restrict__ s,
                                         uint4& d0, uint4& d1, uint4& d2, uint4& d3) {
    const uint4* p = (const uint4*)s;
    d0 = p[0]; d1 = p[1]; d2 = p[2]; d3 = p[3];
}

// ---- LDS fragment loads matching ISA 7.12.2 bf16 layouts ------------------
// A 16x32: lanes 0-15 hold K 0-7 & 16-23 of row M=lane; lanes 16-31 hold K 8-15 & 24-31.
__device__ __forceinline__ void load_fragA(const unsigned short* base, int ldu, int r0, int lane, FragB16& f) {
    int lrow = lane & 15, kh = lane >> 4;
    const unsigned short* p = base + (r0 + lrow) * ldu + kh * 8;
    f.u[0] = *(const uint4*)p;
    f.u[1] = *(const uint4*)(p + 16);
}
// B 32x16: lanes 0-15 hold K 0-7 & 16-23 of col N=lane; lanes 16-31 hold K 8-15 & 24-31.
__device__ __forceinline__ void load_fragB(const unsigned short* base, int ldu, int n0, int lane, FragB16& f) {
    int lrow = lane & 15, kh = lane >> 4;
    const unsigned short* p = base + (n0 + lrow) * ldu + kh * 8;
    f.u[0] = *(const uint4*)p;
    f.u[1] = *(const uint4*)(p + 16);
}

__device__ __forceinline__ void store_out(float* p, float x) { *p = x; }
__device__ __forceinline__ void store_out(unsigned short* p, float x) { *p = f2bf(x); }

// ---------------------------------------------------------------------------
// GEMM: out[m,n] = act( sum_k A[m,k] * W[n,k] + bias[n] )
// A: [M,K] row-major (f32 or bf16), W: [N,K] row-major bf16, out row stride out_ld.
// Block tile 128x256x32, 256 threads = 8 waves (2 waveM x 4 waveN),
// wave tile 64x64 = 4x4 WMMA accumulators. Requires M%128==N%256==K%32==0.
// ---------------------------------------------------------------------------
template <typename AT, typename OT, int ACT>
__global__ __launch_bounds__(256) void gemm_wmma(
    const AT* __restrict__ A, const unsigned short* __restrict__ W,
    const float* __restrict__ bias, OT* __restrict__ out,
    int M, int N, int K, int out_ld)
{
    __shared__ __align__(16) unsigned short sA[2][128 * 40];  // 20 KB
    __shared__ __align__(16) unsigned short sB[2][256 * 40];  // 40 KB

    const int tid   = threadIdx.x;
    const int lane  = tid & 31;
    const int wv    = tid >> 5;
    const int waveM = wv & 1;        // 0..1  (64 rows each)
    const int waveN = wv >> 1;       // 0..3  (64 cols each)
    const int rowBase = blockIdx.y * 128;
    const int colBase = blockIdx.x * 256;

    // A-tile loader: 2 threads per row, 16 elements each
    const int aRow = tid >> 1;           // 0..127
    const int aSeg = (tid & 1) * 16;     // 0 or 16
    // B-tile loader: 1 thread per row, 32 elements each
    const AT* aP             = A + (size_t)(rowBase + aRow) * K + aSeg;
    const unsigned short* wP = W + (size_t)(colBase + tid) * K;

    v8f acc[4][4];
    v8f zero = {0.f,0.f,0.f,0.f,0.f,0.f,0.f,0.f};
#pragma unroll
    for (int i = 0; i < 4; ++i)
#pragma unroll
        for (int j = 0; j < 4; ++j) acc[i][j] = zero;

    const int KT = K >> 5;
    uint4 a0, a1, b0, b1, b2, b3;

    // preload k-tile 0
    fetch16(aP, a0, a1);
    fetchW32(wP, b0, b1, b2, b3);
    {
        uint4* pa = (uint4*)&sA[0][aRow * 40 + aSeg]; pa[0] = a0; pa[1] = a1;
        uint4* pb = (uint4*)&sB[0][tid * 40];
        pb[0] = b0; pb[1] = b1; pb[2] = b2; pb[3] = b3;
    }
    __syncthreads();

    for (int kt = 0; kt < KT; ++kt) {
        const int cur = kt & 1, nxt = cur ^ 1;
        const bool more = (kt + 1 < KT);
        if (more) {
            fetch16(aP + 32, a0, a1);
            fetchW32(wP + 32, b0, b1, b2, b3);
            __builtin_prefetch(aP + 64, 0, 1);   // speculative K+2 prefetch
            __builtin_prefetch(wP + 64, 0, 1);
        }

        FragB16 af[4], bfg[4];
#pragma unroll
        for (int ms = 0; ms < 4; ++ms)
            load_fragA(&sA[cur][0], 40, waveM * 64 + ms * 16, lane, af[ms]);
#pragma unroll
        for (int ns = 0; ns < 4; ++ns)
            load_fragB(&sB[cur][0], 40, waveN * 64 + ns * 16, lane, bfg[ns]);

#pragma unroll
        for (int ms = 0; ms < 4; ++ms)
#pragma unroll
            for (int ns = 0; ns < 4; ++ns)
                acc[ms][ns] = __builtin_amdgcn_wmma_f32_16x16x32_bf16(
                    false, af[ms].v, false, bfg[ns].v,
                    (short)0, acc[ms][ns], false, false);

        __syncthreads();
        if (more) {
            uint4* pa = (uint4*)&sA[nxt][aRow * 40 + aSeg]; pa[0] = a0; pa[1] = a1;
            uint4* pb = (uint4*)&sB[nxt][tid * 40];
            pb[0] = b0; pb[1] = b1; pb[2] = b2; pb[3] = b3;
            __syncthreads();
        }
        aP += 32;
        wP += 32;
    }

    // Epilogue. C/D layout: VGPR v -> M = v + 8*(lane>>4), N = lane&15.
    const int lcol  = lane & 15;
    const int rhalf = (lane >> 4) * 8;
#pragma unroll
    for (int ms = 0; ms < 4; ++ms) {
#pragma unroll
        for (int ns = 0; ns < 4; ++ns) {
            const int col = colBase + waveN * 64 + ns * 16 + lcol;
            const float bv = bias[col];
#pragma unroll
            for (int v = 0; v < 8; ++v) {
                const int row = rowBase + waveM * 64 + ms * 16 + rhalf + v;
                float x = acc[ms][ns][v] + bv;
                if (ACT == 1) x = gelu_exact(x);
                store_out(out + (size_t)row * out_ld + col, x);
            }
        }
    }
}

// ---------------------------------------------------------------------------
// LayerNorm over F=1024, row r = b*2+m selects (ln1, ln2) params, bf16 output.
// ---------------------------------------------------------------------------
__global__ __launch_bounds__(256) void ln_kernel(
    const float* __restrict__ in,
    const float* __restrict__ g1, const float* __restrict__ b1,
    const float* __restrict__ g2, const float* __restrict__ b2,
    unsigned short* __restrict__ outbf)
{
    const int row = blockIdx.x;
    const int tid = threadIdx.x;
    const int lane = tid & 31, wv = tid >> 5;
    const float* gp = (row & 1) ? g2 : g1;
    const float* bp = (row & 1) ? b2 : b1;
    const float* x = in + (size_t)row * 1024;

    float4 v = ((const float4*)x)[tid];
    float s  = v.x + v.y + v.z + v.w;
    float s2 = v.x*v.x + v.y*v.y + v.z*v.z + v.w*v.w;
    for (int off = 16; off; off >>= 1) {
        s  += __shfl_xor(s,  off, 32);
        s2 += __shfl_xor(s2, off, 32);
    }
    __shared__ float ssum[8], ssq[8];
    if (lane == 0) { ssum[wv] = s; ssq[wv] = s2; }
    __syncthreads();
    if (tid == 0) {
        float a = 0.f, b = 0.f;
        for (int i = 0; i < 8; ++i) { a += ssum[i]; b += ssq[i]; }
        ssum[0] = a; ssq[0] = b;
    }
    __syncthreads();
    const float mean = ssum[0] * (1.0f / 1024.0f);
    const float var  = ssq[0] * (1.0f / 1024.0f) - mean * mean;
    const float inv  = rsqrtf(var + 1e-5f);

    float4 g = ((const float4*)gp)[tid];
    float4 b = ((const float4*)bp)[tid];
    float y0 = (v.x - mean) * inv * g.x + b.x;
    float y1 = (v.y - mean) * inv * g.y + b.y;
    float y2 = (v.z - mean) * inv * g.z + b.z;
    float y3 = (v.w - mean) * inv * g.w + b.w;
    uint2 p; p.x = packbf2(y0, y1); p.y = packbf2(y2, y3);
    ((uint2*)(outbf + (size_t)row * 1024))[tid] = p;
}

// ---------------------------------------------------------------------------
// Attention, seq-len 2, HD=128: one wave per (b,h); lane handles 4 dims.
// qkv bf16 [B*2, 3072] (q|k|v), o bf16 [B*2, 1024].
// ---------------------------------------------------------------------------
__device__ __forceinline__ void ld4bf(const unsigned short* p, float* o) {
    uint2 u = *(const uint2*)p;
    o[0] = bf2f((unsigned short)(u.x & 0xffff)); o[1] = bf2f((unsigned short)(u.x >> 16));
    o[2] = bf2f((unsigned short)(u.y & 0xffff)); o[3] = bf2f((unsigned short)(u.y >> 16));
}
__global__ __launch_bounds__(256) void attn_kernel(
    const unsigned short* __restrict__ qkv, unsigned short* __restrict__ o)
{
    const int unit = blockIdx.x * 8 + (threadIdx.x >> 5);   // b*8 + h
    const int lane = threadIdx.x & 31;
    const int b = unit >> 3, h = unit & 7;
    const size_t r0 = (size_t)(b * 2) * 3072;
    const size_t r1 = r0 + 3072;
    const int dbase = h * 128 + lane * 4;

    float q0[4], q1[4], k0[4], k1[4], v0[4], v1[4];
    ld4bf(qkv + r0 + dbase,        q0);
    ld4bf(qkv + r1 + dbase,        q1);
    ld4bf(qkv + r0 + 1024 + dbase, k0);
    ld4bf(qkv + r1 + 1024 + dbase, k1);
    ld4bf(qkv + r0 + 2048 + dbase, v0);
    ld4bf(qkv + r1 + 2048 + dbase, v1);

    float s00 = 0.f, s01 = 0.f, s10 = 0.f, s11 = 0.f;
#pragma unroll
    for (int i = 0; i < 4; ++i) {
        s00 += q0[i] * k0[i]; s01 += q0[i] * k1[i];
        s10 += q1[i] * k0[i]; s11 += q1[i] * k1[i];
    }
    for (int off = 16; off; off >>= 1) {
        s00 += __shfl_xor(s00, off, 32); s01 += __shfl_xor(s01, off, 32);
        s10 += __shfl_xor(s10, off, 32); s11 += __shfl_xor(s11, off, 32);
    }
    const float scale = 0.08838834764831845f;  // 1/sqrt(128)
    s00 *= scale; s01 *= scale; s10 *= scale; s11 *= scale;

    float m0 = fmaxf(s00, s01), m1 = fmaxf(s10, s11);
    float e00 = expf(s00 - m0), e01 = expf(s01 - m0);
    float e10 = expf(s10 - m1), e11 = expf(s11 - m1);
    float a00 = e00 / (e00 + e01), a01 = e01 / (e00 + e01);
    float a10 = e10 / (e10 + e11), a11 = e11 / (e10 + e11);

    unsigned short* o0 = o + (size_t)(b * 2) * 1024 + h * 128 + lane * 4;
    unsigned short* o1 = o0 + 1024;
    uint2 w0, w1;
    w0.x = packbf2(a00 * v0[0] + a01 * v1[0], a00 * v0[1] + a01 * v1[1]);
    w0.y = packbf2(a00 * v0[2] + a01 * v1[2], a00 * v0[3] + a01 * v1[3]);
    w1.x = packbf2(a10 * v0[0] + a11 * v1[0], a10 * v0[1] + a11 * v1[1]);
    w1.y = packbf2(a10 * v0[2] + a11 * v1[2], a10 * v0[3] + a11 * v1[3]);
    *(uint2*)o0 = w0;
    *(uint2*)o1 = w1;
}

// ---------------------------------------------------------------------------
// Gate head: gate_h[b,:]@w2.T + b2 -> softmax(2) -> gate_weights; fuse attn.
// ---------------------------------------------------------------------------
__global__ __launch_bounds__(256) void gate_fuse_kernel(
    const float* __restrict__ gate_h, const float* __restrict__ w2,
    const float* __restrict__ b2, const float* __restrict__ attn_out,
    float* __restrict__ fused_in, float* __restrict__ gate_w_out)
{
    const int b = blockIdx.x;
    const int tid = threadIdx.x;
    const int lane = tid & 31, wv = tid >> 5;

    float4 hv = ((const float4*)(gate_h + (size_t)b * 1024))[tid];
    float4 wa = ((const float4*)w2)[tid];
    float4 wb = ((const float4*)(w2 + 1024))[tid];
    float d0 = hv.x*wa.x + hv.y*wa.y + hv.z*wa.z + hv.w*wa.w;
    float d1 = hv.x*wb.x + hv.y*wb.y + hv.z*wb.z + hv.w*wb.w;
    for (int off = 16; off; off >>= 1) {
        d0 += __shfl_xor(d0, off, 32);
        d1 += __shfl_xor(d1, off, 32);
    }
    __shared__ float r0[8], r1[8], gw[2];
    if (lane == 0) { r0[wv] = d0; r1[wv] = d1; }
    __syncthreads();
    if (tid == 0) {
        float l0 = b2[0], l1 = b2[1];
        for (int i = 0; i < 8; ++i) { l0 += r0[i]; l1 += r1[i]; }
        float m = fmaxf(l0, l1);
        float e0 = expf(l0 - m), e1 = expf(l1 - m);
        float inv = 1.0f / (e0 + e1);
        gw[0] = e0 * inv; gw[1] = e1 * inv;
        gate_w_out[(size_t)b * 2 + 0] = gw[0];
        gate_w_out[(size_t)b * 2 + 1] = gw[1];
    }
    __syncthreads();
    const float g0 = gw[0], g1 = gw[1];
    float4 cv = ((const float4*)(attn_out + (size_t)(b * 2)     * 1024))[tid];
    float4 gv = ((const float4*)(attn_out + (size_t)(b * 2 + 1) * 1024))[tid];
    float4 r;
    r.x = g0 * cv.x + g1 * gv.x; r.y = g0 * cv.y + g1 * gv.y;
    r.z = g0 * cv.z + g1 * gv.z; r.w = g0 * cv.w + g1 * gv.w;
    ((float4*)(fused_in + (size_t)b * 1024))[tid] = r;
}

// ---------------------------------------------------------------------------
// Weight f32 -> bf16 (n % 4 == 0)
// ---------------------------------------------------------------------------
__global__ __launch_bounds__(256) void cvt_bf16_kernel(
    const float* __restrict__ s, unsigned short* __restrict__ d, int n4)
{
    int i = blockIdx.x * 256 + threadIdx.x;
    if (i < n4) {
        float4 v = ((const float4*)s)[i];
        uint2 p; p.x = packbf2(v.x, v.y); p.y = packbf2(v.z, v.w);
        ((uint2*)d)[i] = p;
    }
}

// ---------------------------------------------------------------------------
extern "C" void kernel_launch(void* const* d_in, const int* in_sizes, int n_in,
                              void* d_out, int out_size, void* d_ws, size_t ws_size,
                              hipStream_t stream) {
    (void)in_sizes; (void)n_in; (void)out_size; (void)ws_size;
    constexpr int B = 8192, CD = 512, F = 1024;

    const float* cnn_embed  = (const float*)d_in[0];
    const float* gnn_embed  = (const float*)d_in[1];
    const float* cnn_w      = (const float*)d_in[2];
    const float* cnn_b      = (const float*)d_in[3];
    const float* gnn_w      = (const float*)d_in[4];
    const float* gnn_b      = (const float*)d_in[5];
    const float* ln1_g      = (const float*)d_in[6];
    const float* ln1_b      = (const float*)d_in[7];
    const float* ln2_g      = (const float*)d_in[8];
    const float* ln2_b      = (const float*)d_in[9];
    const float* attn_in_w  = (const float*)d_in[10];
    const float* attn_in_b  = (const float*)d_in[11];
    const float* attn_out_w = (const float*)d_in[12];
    const float* attn_out_b = (const float*)d_in[13];
    const float* gate_w1    = (const float*)d_in[14];
    const float* gate_b1    = (const float*)d_in[15];
    const float* gate_w2    = (const float*)d_in[16];
    const float* gate_b2    = (const float*)d_in[17];
    const float* proj_w     = (const float*)d_in[18];
    const float* proj_b     = (const float*)d_in[19];

    char* ws = (char*)d_ws;
    size_t off = 0;
    auto alloc = [&](size_t bytes) -> void* {
        off = (off + 255) & ~(size_t)255;
        void* p = ws + off;
        off += bytes;
        return p;
    };

    // bf16 weights
    unsigned short* wb_cnn   = (unsigned short*)alloc((size_t)F * CD * 2);
    unsigned short* wb_gnn   = (unsigned short*)alloc((size_t)F * CD * 2);
    unsigned short* wb_qkv   = (unsigned short*)alloc((size_t)3 * F * F * 2);
    unsigned short* wb_ao    = (unsigned short*)alloc((size_t)F * F * 2);
    unsigned short* wb_g1    = (unsigned short*)alloc((size_t)F * 2 * F * 2);
    unsigned short* wb_proj  = (unsigned short*)alloc((size_t)F * F * 2);

    // activation regions (aliased across phases; one stream serializes use)
    float* tmp_cg = (float*)alloc((size_t)B * 2 * F * 4);            // pre-LN; later gate_h + fused_in
    unsigned short* combined = (unsigned short*)alloc((size_t)B * 2 * F * 2);  // later o
    unsigned short* qkv = (unsigned short*)alloc((size_t)B * 2 * 3 * F * 2);   // later attn_out

    float* gate_h   = tmp_cg;                      // [B, F] f32 (tmp_cg dead after LN)
    float* fused_in = tmp_cg + (size_t)B * F;      // [B, F] f32
    unsigned short* obuf = combined;               // [B*2, F] bf16 (combined dead after qkv GEMM)
    float* attn_out = (float*)qkv;                 // [B*2, F] f32 (qkv dead after attention)

    float* fused_out = (float*)d_out;              // [B, F]
    float* gate_wout = fused_out + (size_t)B * F;  // [B, 2]

    // 1) convert weights to bf16 (amortized; weights fit in 192MB L2)
    auto cvt = [&](const float* s, unsigned short* d, int n) {
        int n4 = n / 4;
        cvt_bf16_kernel<<<(n4 + 255) / 256, 256, 0, stream>>>(s, d, n4);
    };
    cvt(cnn_w,      wb_cnn,  F * CD);
    cvt(gnn_w,      wb_gnn,  F * CD);
    cvt(attn_in_w,  wb_qkv,  3 * F * F);
    cvt(attn_out_w, wb_ao,   F * F);
    cvt(gate_w1,    wb_g1,   F * 2 * F);
    cvt(proj_w,     wb_proj, F * F);

    // 2) embedding projections (interleaved output rows b*2 + m)
    {
        dim3 grid(F / 256, B / 128);
        gemm_wmma<float, float, 0><<<grid, 256, 0, stream>>>(
            cnn_embed, wb_cnn, cnn_b, tmp_cg, B, F, CD, 2 * F);
        gemm_wmma<float, float, 0><<<grid, 256, 0, stream>>>(
            gnn_embed, wb_gnn, gnn_b, tmp_cg + F, B, F, CD, 2 * F);
    }

    // 3) layernorm -> bf16 combined [B*2, F]
    ln_kernel<<<B * 2, 256, 0, stream>>>(tmp_cg, ln1_g, ln1_b, ln2_g, ln2_b, combined);

    // 4) qkv GEMM [16384 x 3072 x 1024] -> bf16
    {
        dim3 grid(3 * F / 256, 2 * B / 128);
        gemm_wmma<unsigned short, unsigned short, 0><<<grid, 256, 0, stream>>>(
            combined, wb_qkv, attn_in_b, qkv, 2 * B, 3 * F, F, 3 * F);
    }

    // 5) 2-token attention -> o bf16 (overwrites combined region)
    attn_kernel<<<(B * 8) / 8, 256, 0, stream>>>(qkv, obuf);

    // 6) attn_out GEMM -> f32 (overwrites qkv region)
    {
        dim3 grid(F / 256, 2 * B / 128);
        gemm_wmma<unsigned short, float, 0><<<grid, 256, 0, stream>>>(
            obuf, wb_ao, attn_out_b, attn_out, 2 * B, F, F, F);
    }

    // 7) gate hidden: attn_out viewed as [B, 2F] == concat(c_attn, g_attn); GELU
    {
        dim3 grid(F / 256, B / 128);
        gemm_wmma<float, float, 1><<<grid, 256, 0, stream>>>(
            attn_out, wb_g1, gate_b1, gate_h, B, F, 2 * F, F);
    }

    // 8) gate softmax + fusion
    gate_fuse_kernel<<<B, 256, 0, stream>>>(gate_h, gate_w2, gate_b2,
                                            attn_out, fused_in, gate_wout);

    // 9) final projection + GELU -> d_out
    {
        dim3 grid(F / 256, B / 128);
        gemm_wmma<float, float, 1><<<grid, 256, 0, stream>>>(
            fused_in, wb_proj, proj_b, fused_out, B, F, F, F);
    }
}
